// RSSM_1941325218274
// MI455X (gfx1250) — matrix operand order repaired and checked by reading, not compiled, not run on hardware
//
#include <hip/hip_runtime.h>
#include <hip/hip_bf16.h>
#include <math.h>

typedef _Float16 f16;
typedef __attribute__((ext_vector_type(16))) _Float16 v16h;
typedef __attribute__((ext_vector_type(8)))  _Float16 v8h;
typedef __attribute__((ext_vector_type(8)))  float    v8f;

namespace {
constexpr int B = 256, T = 256, E = 256, D = 1024, S = 256, H = 512;
constexpr float MIN_STD = 0.1f;
constexpr int BT = 16;             // batch rows per workgroup
constexpr int NWAVE = 16;          // waves per workgroup
constexpr int BLOCK = NWAVE * 32;  // 512 threads

// Packed f16 weight layout in d_ws: fragment = 32 lanes x 16 halves = 512 elem.
constexpr size_t FRAG = 512;
constexpr size_t W_IH_OFF = 0;
constexpr size_t W_IH_SZ  = (size_t)3 * D * S;
constexpr size_t W_HH_OFF = W_IH_OFF + W_IH_SZ;
constexpr size_t W_HH_SZ  = (size_t)3 * D * D;
constexpr size_t W_P1_OFF = W_HH_OFF + W_HH_SZ;
constexpr size_t W_P1_SZ  = (size_t)H * D;
constexpr size_t W_P2_OFF = W_P1_OFF + W_P1_SZ;
constexpr size_t W_P2_SZ  = (size_t)2 * S * H;
constexpr size_t W_Q1_OFF = W_P2_OFF + W_P2_SZ;
constexpr size_t W_Q1_SZ  = (size_t)H * (D + E);
constexpr size_t W_Q2_OFF = W_Q1_OFF + W_Q1_SZ;
} // namespace

// Convert f32 weight (N x K, row-major, used as B = W^T) into fragment-native
// f16 tiles. Fragment f = nt*(K/32)+kt; within fragment: lane l, elem i maps to
// B-operand layout: n = l&15, k = kt*32 + (l>>4)*8 + (i&7) + (i>>3)*16.
__global__ void pack_weight(const float* __restrict__ src, f16* __restrict__ dst,
                            int N, int K) {
  int o = blockIdx.x * blockDim.x + threadIdx.x;
  if (o >= N * K) return;
  int i  = o & 15;
  int l  = (o >> 4) & 31;
  int fi = o >> 9;
  int ktiles = K >> 5;
  int kt = fi % ktiles;
  int nt = fi / ktiles;
  int row = nt * 16 + (l & 15);
  int col = kt * 32 + ((l >> 4) << 3) + (i & 7) + ((i >> 3) << 4);
  dst[o] = (f16)src[(size_t)row * K + col];
}

__device__ __forceinline__ v8f wmma32(v16h a, v16h b, v8f c) {
  return __builtin_amdgcn_wmma_f32_16x16x32_f16(false, a, false, b,
                                                (short)0, c, false, false);
}

// A fragment (16x32 f16) from LDS row-major [16 x ldk], tile at column k0.
// CDNA5 layout: lane m=l&15 holds row m; lanes 0-15 carry K {0..7,16..23},
// lanes 16-31 carry K {8..15,24..31}. Two 16B LDS loads per lane.
__device__ __forceinline__ v16h ldA(const f16* base, int ldk, int lane, int k0) {
  int m  = lane & 15;
  int kb = (lane >> 4) << 3;
  const f16* p = base + m * ldk + k0 + kb;
  v8h lo = *(const v8h*)(p);
  v8h hi = *(const v8h*)(p + 16);
  v16h r;
#pragma unroll
  for (int i = 0; i < 8; ++i) { r[i] = lo[i]; r[8 + i] = hi[i]; }
  return r;
}

// B fragment from packed global weights: one 32B load per lane.
__device__ __forceinline__ v16h ldB(const f16* frag, int lane) {
  return *(const v16h*)(frag + lane * 16);
}

__device__ __forceinline__ float sigmoidf_(float x) {
  return 1.0f / (1.0f + __expf(-x));
}
__device__ __forceinline__ float softplusf_(float x) {
  return (x > 20.0f) ? x : log1pf(__expf(x));
}

__global__ __launch_bounds__(BLOCK, 1)
void rssm_kernel(const float* __restrict__ embeds, const float* __restrict__ noise,
                 const float* __restrict__ h0, const float* __restrict__ z0,
                 const float* __restrict__ bih, const float* __restrict__ bhh,
                 const float* __restrict__ pb1, const float* __restrict__ pb2,
                 const float* __restrict__ qb1, const float* __restrict__ qb2,
                 const f16* __restrict__ wpack, float* __restrict__ out) {
  extern __shared__ __align__(16) char smem_raw[];
  f16* hbuf   = (f16*)smem_raw;            // 2 * 16 * D (double-buffered h, f16)
  f16* z_lds  = hbuf + 2 * BT * D;         // 16 * S
  f16* e_lds  = z_lds + BT * S;            // 16 * E
  f16* hp_lds = e_lds + BT * E;            // 16 * H (prior hidden)
  f16* hq_lds = hp_lds + BT * H;           // 16 * H (posterior hidden)

  const int tid  = threadIdx.x;
  const int lane = tid & 31;
  const int w    = tid >> 5;               // wave id 0..15
  const int b0   = blockIdx.x * BT;
  const int n    = lane & 15;              // N index within 16x16 tile
  const int mb   = (lane >> 4) << 3;       // row base for C/D fragment

  const f16* wih = wpack + W_IH_OFF;
  const f16* whh = wpack + W_HH_OFF;
  const f16* wp1 = wpack + W_P1_OFF;
  const f16* wp2 = wpack + W_P2_OFF;
  const f16* wq1 = wpack + W_Q1_OFF;
  const f16* wq2 = wpack + W_Q2_OFF;

  float* out_h  = out;
  float* out_z  = out + (size_t)B * T * D;
  float* out_pm = out_z  + (size_t)B * T * S;
  float* out_ps = out_pm + (size_t)B * T * S;
  float* out_qm = out_ps + (size_t)B * T * S;
  float* out_qs = out_qm + (size_t)B * T * S;

  // Master f32 copy of h for the columns this wave owns (4 d-groups x 8 rows).
  float hreg[4][8];
#pragma unroll
  for (int g = 0; g < 4; ++g) {
    int dg = w + NWAVE * g;
#pragma unroll
    for (int j = 0; j < 8; ++j)
      hreg[g][j] = h0[(size_t)(b0 + mb + j) * D + dg * 16 + n];
  }
  for (int i = tid; i < BT * D; i += BLOCK)
    hbuf[i] = (f16)h0[(size_t)(b0 + (i >> 10)) * D + (i & (D - 1))];
  for (int i = tid; i < BT * S; i += BLOCK)
    z_lds[i] = (f16)z0[(size_t)(b0 + (i >> 8)) * S + (i & (S - 1))];
  __syncthreads();

  for (int t = 0; t < T; ++t) {
    const f16* hc = hbuf + (t & 1) * BT * D;       // h_{t-1}
    f16* hnx      = hbuf + ((t + 1) & 1) * BT * D; // h_t

    // Stage e_t (f32 -> f16) into LDS; consumed after barrier A.
    for (int i = tid; i < BT * E; i += BLOCK) {
      int m = i >> 8, c = i & (E - 1);
      e_lds[i] = (f16)embeds[((size_t)(b0 + m) * T + t) * E + c];
    }

    // ---------------- GRU: each wave owns d-groups {w, w+16, w+32, w+48} ----
#pragma unroll
    for (int g = 0; g < 4; ++g) {
      int dg = w + NWAVE * g;
      v8f ar = {}, au = {}, agin = {}, aghn = {};
      const f16* br = wih + (size_t)dg          * (S / 32) * FRAG;
      const f16* bu = wih + (size_t)(64 + dg)   * (S / 32) * FRAG;
      const f16* bn = wih + (size_t)(128 + dg)  * (S / 32) * FRAG;
#pragma unroll 2
      for (int kt = 0; kt < S / 32; ++kt) {       // gi = z @ Wih^T
        v16h a = ldA(z_lds, S, lane, kt * 32);
        ar   = wmma32(a, ldB(br + kt * FRAG, lane), ar);
        au   = wmma32(a, ldB(bu + kt * FRAG, lane), au);
        agin = wmma32(a, ldB(bn + kt * FRAG, lane), agin);
      }
      const f16* cr = whh + (size_t)dg          * (D / 32) * FRAG;
      const f16* cu = whh + (size_t)(64 + dg)   * (D / 32) * FRAG;
      const f16* cn = whh + (size_t)(128 + dg)  * (D / 32) * FRAG;
#pragma unroll 2
      for (int kt = 0; kt < D / 32; ++kt) {       // gh = h @ Whh^T
        v16h a = ldA(hc, D, lane, kt * 32);
        ar   = wmma32(a, ldB(cr + kt * FRAG, lane), ar);   // r: gi+gh fused
        au   = wmma32(a, ldB(cu + kt * FRAG, lane), au);   // u: gi+gh fused
        aghn = wmma32(a, ldB(cn + kt * FRAG, lane), aghn); // n-gate h part
      }
      int d = dg * 16 + n;
      float b_ir = bih[d] + bhh[d];
      float b_iu = bih[D + d] + bhh[D + d];
      float b_in = bih[2 * D + d];
      float b_hn = bhh[2 * D + d];
#pragma unroll
      for (int j = 0; j < 8; ++j) {
        int m = mb + j;
        float r  = sigmoidf_(ar[j] + b_ir);
        float u  = sigmoidf_(au[j] + b_iu);
        float nn = tanhf(agin[j] + b_in + r * (aghn[j] + b_hn));
        float hv = (1.0f - u) * nn + u * hreg[g][j];
        hreg[g][j] = hv;
        hnx[m * D + d] = (f16)hv;
        out_h[((size_t)(b0 + m) * T + t) * D + d] = hv;
      }
    }
    __syncthreads();  // A: h_t and e_t staged

    // ---------------- MLP layer 1 (prior K=D, posterior K=D+E), ELU --------
#pragma unroll 1
    for (int p = 0; p < 2; ++p) {
      int nt = w + NWAVE * p;
      v8f acc = {};
      const f16* bw = wp1 + (size_t)nt * (D / 32) * FRAG;
#pragma unroll 2
      for (int kt = 0; kt < D / 32; ++kt)
        acc = wmma32(ldA(hnx, D, lane, kt * 32), ldB(bw + kt * FRAG, lane), acc);
      int n0 = nt * 16;
      float bv = pb1[n0 + n];
#pragma unroll
      for (int j = 0; j < 8; ++j) {
        float v = acc[j] + bv;
        hp_lds[(mb + j) * H + n0 + n] = (f16)((v > 0.0f) ? v : (__expf(v) - 1.0f));
      }
    }
#pragma unroll 1
    for (int p = 0; p < 2; ++p) {
      int nt = w + NWAVE * p;
      v8f acc = {};
      const f16* bw = wq1 + (size_t)nt * ((D + E) / 32) * FRAG;
#pragma unroll 2
      for (int kt = 0; kt < (D + E) / 32; ++kt) {
        v16h a = (kt < D / 32) ? ldA(hnx, D, lane, kt * 32)
                               : ldA(e_lds, E, lane, (kt - D / 32) * 32);
        acc = wmma32(a, ldB(bw + kt * FRAG, lane), acc);
      }
      int n0 = nt * 16;
      float bv = qb1[n0 + n];
#pragma unroll
      for (int j = 0; j < 8; ++j) {
        float v = acc[j] + bv;
        hq_lds[(mb + j) * H + n0 + n] = (f16)((v > 0.0f) ? v : (__expf(v) - 1.0f));
      }
    }
    __syncthreads();  // B: hidden activations ready

    // ---------------- MLP layer 2: wave w owns s-group w (mu+std paired) ---
    {
      int s0 = w * 16;
      v8f am = {}, as = {};
      const f16* bm = wp2 + (size_t)w        * (H / 32) * FRAG;
      const f16* bs = wp2 + (size_t)(16 + w) * (H / 32) * FRAG;
#pragma unroll 2
      for (int kt = 0; kt < H / 32; ++kt) {
        v16h a = ldA(hp_lds, H, lane, kt * 32);
        am = wmma32(a, ldB(bm + kt * FRAG, lane), am);
        as = wmma32(a, ldB(bs + kt * FRAG, lane), as);
      }
      float bmv = pb2[s0 + n], bsv = pb2[S + s0 + n];
#pragma unroll
      for (int j = 0; j < 8; ++j) {
        size_t idx = ((size_t)(b0 + mb + j) * T + t) * S + s0 + n;
        out_pm[idx] = am[j] + bmv;
        out_ps[idx] = softplusf_(as[j] + bsv) + MIN_STD;
      }
      v8f qm = {}, qs = {};
      const f16* cm = wq2 + (size_t)w        * (H / 32) * FRAG;
      const f16* cs = wq2 + (size_t)(16 + w) * (H / 32) * FRAG;
#pragma unroll 2
      for (int kt = 0; kt < H / 32; ++kt) {
        v16h a = ldA(hq_lds, H, lane, kt * 32);
        qm = wmma32(a, ldB(cm + kt * FRAG, lane), qm);
        qs = wmma32(a, ldB(cs + kt * FRAG, lane), qs);
      }
      float cmv = qb2[s0 + n], csv = qb2[S + s0 + n];
#pragma unroll
      for (int j = 0; j < 8; ++j) {
        int m = mb + j;
        size_t idx = ((size_t)(b0 + m) * T + t) * S + s0 + n;
        float mu = qm[j] + cmv;
        float sd = softplusf_(qs[j] + csv) + MIN_STD;
        float zz = mu + sd * noise[idx];
        out_qm[idx] = mu;
        out_qs[idx] = sd;
        out_z[idx]  = zz;
        z_lds[m * S + s0 + n] = (f16)zz;  // recurrent sample for t+1
      }
    }
    __syncthreads();  // C: end of step (protects z/e/hidden reuse)
  }
}

extern "C" void kernel_launch(void* const* d_in, const int* in_sizes, int n_in,
                              void* d_out, int out_size, void* d_ws, size_t ws_size,
                              hipStream_t stream) {
  (void)in_sizes; (void)n_in; (void)out_size; (void)ws_size;
  const float* embeds = (const float*)d_in[0];
  const float* noise  = (const float*)d_in[1];
  const float* h0     = (const float*)d_in[2];
  const float* z0     = (const float*)d_in[3];
  const float* wih    = (const float*)d_in[4];
  const float* whh    = (const float*)d_in[5];
  const float* bih    = (const float*)d_in[6];
  const float* bhh    = (const float*)d_in[7];
  const float* pw1    = (const float*)d_in[8];
  const float* pb1    = (const float*)d_in[9];
  const float* pw2    = (const float*)d_in[10];
  const float* pb2    = (const float*)d_in[11];
  const float* qw1    = (const float*)d_in[12];
  const float* qb1    = (const float*)d_in[13];
  const float* qw2    = (const float*)d_in[14];
  const float* qb2    = (const float*)d_in[15];
  f16* wp = (f16*)d_ws;

  auto pack = [&](const float* src, size_t off, int N, int K) {
    int total = N * K;
    pack_weight<<<(total + 255) / 256, 256, 0, stream>>>(src, wp + off, N, K);
  };
  pack(wih, W_IH_OFF, 3 * D, S);
  pack(whh, W_HH_OFF, 3 * D, D);
  pack(pw1, W_P1_OFF, H, D);
  pack(pw2, W_P2_OFF, 2 * S, H);
  pack(qw1, W_Q1_OFF, H, D + E);
  pack(qw2, W_Q2_OFF, 2 * S, H);

  size_t smem = (size_t)(2 * BT * D + BT * S + BT * E + 2 * BT * H) * sizeof(f16);
  rssm_kernel<<<B / BT, BLOCK, smem, stream>>>(embeds, noise, h0, z0,
                                               bih, bhh, pb1, pb2, qb1, qb2,
                                               wp, (float*)d_out);
}
